// GatedGCNNet_40699110097233
// MI455X (gfx1250) — matrix operand order repaired and checked by reading, not compiled, not run on hardware
//
#include <hip/hip_runtime.h>
#include <hip/hip_bf16.h>

typedef __attribute__((ext_vector_type(16))) _Float16 v16h;
typedef __attribute__((ext_vector_type(8)))  float    v8f;

#define NNODES 50000
#define NEDGES 1000000
#define IN_DIM 128
#define HID    64
#define NLAYER 4
#define NH     (NNODES * HID)          // 3,200,000 elements per node matrix

#define EMB_PACK (4 * 4 * 32 * 16)     // 8192 halves: 4 ks-steps x 4 n-tiles x 32 lanes x 16 halves
#define MAT_PACK (2 * 4 * 32 * 16)     // 4096 halves: 2 ks-steps x 4 n-tiles
#define PACK_TOTAL (EMB_PACK + NLAYER * 4 * MAT_PACK)   // 73728 halves = 147456 B

// ---------------------------------------------------------------------------
// WMMA fragment layouts (CDNA5 ISA 7.12.2, wave32)
//  A 16x32 f16: lane l -> row M=l%16; half j -> K=(l/16)*16+j (contiguous order, as load_a_frag builds it)
//  B 32x16 f16: lane l -> col N=l%16; half j -> K=(l/16)*16+j (this code's order, same as A; not the ISA table order)
//  C/D 16x16 f32: lane l -> col N=l%16; vgpr r -> row M=(l/16)*8+r
// ---------------------------------------------------------------------------
__device__ __forceinline__ v16h load_a_frag(const float* __restrict__ row, int grp, int k0) {
    v16h a;
#pragma unroll
    for (int j = 0; j < 16; ++j) {
        int k = grp * 16 + j;
        a[j] = (_Float16)row[k0 + k];
    }
    return a;
}

// Packed B fragment: contiguous 16 halves per lane, fragment index f = ks*4 + nt
__device__ __forceinline__ v16h load_b_packed(const _Float16* __restrict__ Wp, int frag, int lane) {
    return *(const v16h*)(Wp + ((size_t)frag * 32 + lane) * 16);
}

__device__ __forceinline__ v8f wmma_f16(v16h a, v16h b, v8f c) {
    return __builtin_amdgcn_wmma_f32_16x16x32_f16(false, a, false, b, (short)0, c, false, false);
}

// ---------------------------------------------------------------------------
// Pack all weights (emb + 4 layers x {A,B,D,E}) into f16 B-fragment layout.
// Packed element (frag, lane, j) = W[ks*32 + (lane/16)*16 + j][nt*16 + lane%16]
// ---------------------------------------------------------------------------
__global__ __launch_bounds__(256) void pack_weights(const float* __restrict__ emb_w,
                                                    const float* __restrict__ Aw,
                                                    const float* __restrict__ Bw,
                                                    const float* __restrict__ Dw,
                                                    const float* __restrict__ Ew,
                                                    _Float16* __restrict__ out) {
    int i = blockIdx.x * blockDim.x + threadIdx.x;
    if (i >= PACK_TOTAL) return;

    const float* W;
    int o;
    if (i < EMB_PACK) {
        W = emb_w;
        o = i;
    } else {
        int t   = i - EMB_PACK;
        int mat = t / MAT_PACK;            // layer*4 + {0:A,1:B,2:D,3:E}
        o       = t - mat * MAT_PACK;
        int l   = mat >> 2, wsel = mat & 3;
        const float* base = (wsel == 0) ? Aw : (wsel == 1) ? Bw : (wsel == 2) ? Dw : Ew;
        W = base + (size_t)l * HID * HID;
    }
    int frag = o >> 9;                     // 512 halves per fragment
    int w    = o & 511;
    int lane = w >> 4;
    int j    = w & 15;
    int ks   = frag >> 2;
    int nt   = frag & 3;
    int k    = ks * 32 + (lane >> 4) * 16 + j;
    int n    = nt * 16 + (lane & 15);
    out[i] = (_Float16)W[k * HID + n];
}

// ---------------------------------------------------------------------------
// Embedding GEMM: Y[N,64] = X[N,128] @ W[128,64] + bias
// ---------------------------------------------------------------------------
__global__ __launch_bounds__(256) void embed_gemm(const float* __restrict__ X,
                                                  const _Float16* __restrict__ Wp,
                                                  const float* __restrict__ bias,
                                                  float* __restrict__ Y) {
    int wave = (blockIdx.x * blockDim.x + threadIdx.x) >> 5;
    int lane = threadIdx.x & 31;
    int row0 = wave * 16;
    if (row0 >= NNODES) return;
    int grp = lane >> 4, m = lane & 15, nlo = lane & 15;

    const float* Arow = X + (size_t)(row0 + m) * IN_DIM;
    v16h afr[4];
#pragma unroll
    for (int ks = 0; ks < 4; ++ks) afr[ks] = load_a_frag(Arow, grp, ks * 32);

#pragma unroll
    for (int nt = 0; nt < 4; ++nt) {
        v8f c = {};
#pragma unroll
        for (int ks = 0; ks < 4; ++ks)
            c = wmma_f16(afr[ks], load_b_packed(Wp, ks * 4 + nt, lane), c);
        float bv = bias[nt * 16 + nlo];
#pragma unroll
        for (int r = 0; r < 8; ++r)
            Y[(size_t)(row0 + grp * 8 + r) * HID + nt * 16 + nlo] = c[r] + bv;
    }
}

// ---------------------------------------------------------------------------
// Fused per-layer node transforms: Ah,Bh,Dh,Eh = h @ {Aw,Bw,Dw,Ew} + bias
// A fragments loaded once, reused for all 4 weight matrices
// ---------------------------------------------------------------------------
__global__ __launch_bounds__(256) void layer_gemms(const float* __restrict__ h,
                                                   const _Float16* __restrict__ Wpack,
                                                   const float* __restrict__ ba,
                                                   const float* __restrict__ bb,
                                                   const float* __restrict__ bd,
                                                   const float* __restrict__ be,
                                                   float* __restrict__ Ah, float* __restrict__ Bh,
                                                   float* __restrict__ Dh, float* __restrict__ Eh) {
    int wave = (blockIdx.x * blockDim.x + threadIdx.x) >> 5;
    int lane = threadIdx.x & 31;
    int row0 = wave * 16;
    if (row0 >= NNODES) return;
    int grp = lane >> 4, m = lane & 15, nlo = lane & 15;

    const float* Arow = h + (size_t)(row0 + m) * HID;
    v16h a0 = load_a_frag(Arow, grp, 0);
    v16h a1 = load_a_frag(Arow, grp, 32);

    const float* bs[4]   = {ba, bb, bd, be};
    float*       outs[4] = {Ah, Bh, Dh, Eh};

#pragma unroll
    for (int w = 0; w < 4; ++w) {
        const _Float16* Wp = Wpack + (size_t)w * MAT_PACK;
        float*          O  = outs[w];
#pragma unroll
        for (int nt = 0; nt < 4; ++nt) {
            int n = nt * 16 + nlo;
            v8f c = {};
            c = wmma_f16(a0, load_b_packed(Wp, 0 * 4 + nt, lane), c);
            c = wmma_f16(a1, load_b_packed(Wp, 1 * 4 + nt, lane), c);
            float bv = bs[w][n];
#pragma unroll
            for (int r = 0; r < 8; ++r)
                O[(size_t)(row0 + grp * 8 + r) * HID + n] = c[r] + bv;
        }
    }
}

// ---------------------------------------------------------------------------
// Edge scatter: one wave per edge, 2 columns per lane (wave32 * 2 = 64)
// sigma = sigmoid(Dh[src] + Eh[dst]); num[dst] += sigma*Bh[src]; den[dst] += sigma
// ---------------------------------------------------------------------------
__global__ __launch_bounds__(256) void edge_scatter(const long long* __restrict__ ei,
                                                    const float* __restrict__ Bh,
                                                    const float* __restrict__ Dh,
                                                    const float* __restrict__ Eh,
                                                    float* __restrict__ num,
                                                    float* __restrict__ den) {
    int e = (blockIdx.x * blockDim.x + threadIdx.x) >> 5;
    if (e >= NEDGES) return;
    int lane = threadIdx.x & 31;
    long long s = ei[e];            // uniform per wave -> scalar load
    long long d = ei[NEDGES + e];

    const float2 x = ((const float2*)(Dh + (size_t)s * HID))[lane];
    const float2 y = ((const float2*)(Eh + (size_t)d * HID))[lane];
    const float2 b = ((const float2*)(Bh + (size_t)s * HID))[lane];

    float s0 = 1.0f / (1.0f + __expf(-(x.x + y.x)));
    float s1 = 1.0f / (1.0f + __expf(-(x.y + y.y)));

    float* np = num + (size_t)d * HID + lane * 2;
    float* dp = den + (size_t)d * HID + lane * 2;
    __hip_atomic_fetch_add(np,     s0 * b.x, __ATOMIC_RELAXED, __HIP_MEMORY_SCOPE_AGENT);
    __hip_atomic_fetch_add(np + 1, s1 * b.y, __ATOMIC_RELAXED, __HIP_MEMORY_SCOPE_AGENT);
    __hip_atomic_fetch_add(dp,     s0,       __ATOMIC_RELAXED, __HIP_MEMORY_SCOPE_AGENT);
    __hip_atomic_fetch_add(dp + 1, s1,       __ATOMIC_RELAXED, __HIP_MEMORY_SCOPE_AGENT);
}

// ---------------------------------------------------------------------------
// Zero fill
// ---------------------------------------------------------------------------
__global__ __launch_bounds__(256) void zero_f32(float* __restrict__ p, int n) {
    int i = blockIdx.x * blockDim.x + threadIdx.x;
    if (i < n) p[i] = 0.0f;
}

// ---------------------------------------------------------------------------
// BN pass 1: t = Ah + num/(den+1e-6); store t; accumulate per-column sum/sumsq
// ---------------------------------------------------------------------------
__global__ __launch_bounds__(256) void bn_stats(const float* __restrict__ Ah,
                                                const float* __restrict__ num,
                                                const float* __restrict__ den,
                                                float* __restrict__ tmp,
                                                float* __restrict__ stats) {
    __shared__ float ssum[HID];
    __shared__ float ssq[HID];
    if (threadIdx.x < HID) { ssum[threadIdx.x] = 0.0f; ssq[threadIdx.x] = 0.0f; }
    __syncthreads();

    size_t stride = (size_t)gridDim.x * blockDim.x;
    for (size_t i = (size_t)blockIdx.x * blockDim.x + threadIdx.x; i < (size_t)NH; i += stride) {
        float t = Ah[i] + num[i] / (den[i] + 1e-6f);
        tmp[i] = t;
        int c = (int)(i & (HID - 1));
        atomicAdd(&ssum[c], t);
        atomicAdd(&ssq[c], t * t);
    }
    __syncthreads();
    if (threadIdx.x < HID) {
        __hip_atomic_fetch_add(&stats[threadIdx.x],       ssum[threadIdx.x], __ATOMIC_RELAXED, __HIP_MEMORY_SCOPE_AGENT);
        __hip_atomic_fetch_add(&stats[HID + threadIdx.x], ssq[threadIdx.x],  __ATOMIC_RELAXED, __HIP_MEMORY_SCOPE_AGENT);
    }
}

// ---------------------------------------------------------------------------
// BN pass 2: out = h_in + relu(gamma*(t-mu)*rsqrt(var+1e-5)+beta)
// ---------------------------------------------------------------------------
__global__ __launch_bounds__(256) void bn_apply(const float* __restrict__ hin,
                                                const float* __restrict__ tmp,
                                                const float* __restrict__ stats,
                                                const float* __restrict__ gamma,
                                                const float* __restrict__ beta,
                                                float* __restrict__ out) {
    size_t i = (size_t)blockIdx.x * blockDim.x + threadIdx.x;
    if (i >= (size_t)NH) return;
    int c = (int)(i & (HID - 1));
    const float invN = 1.0f / (float)NNODES;
    float mu  = stats[c] * invN;
    float var = stats[HID + c] * invN - mu * mu;
    float t = gamma[c] * (tmp[i] - mu) * rsqrtf(var + 1e-5f) + beta[c];
    out[i] = hin[i] + fmaxf(t, 0.0f);
}

// ---------------------------------------------------------------------------
extern "C" void kernel_launch(void* const* d_in, const int* in_sizes, int n_in,
                              void* d_out, int out_size, void* d_ws, size_t ws_size,
                              hipStream_t stream) {
    const float*      feature = (const float*)d_in[0];
    const long long*  ei      = (const long long*)d_in[1];   // int64 per reference
    const float*      emb_w   = (const float*)d_in[2];
    const float*      emb_b   = (const float*)d_in[3];
    const float*      Aw      = (const float*)d_in[4];
    const float*      Ab      = (const float*)d_in[5];
    const float*      Bw      = (const float*)d_in[6];
    const float*      Bb      = (const float*)d_in[7];
    const float*      Dw      = (const float*)d_in[8];
    const float*      Db      = (const float*)d_in[9];
    const float*      Ew      = (const float*)d_in[10];
    const float*      Eb      = (const float*)d_in[11];
    const float*      gamma   = (const float*)d_in[12];
    const float*      beta    = (const float*)d_in[13];

    // Workspace layout (packed weights first for 32B-aligned v16h loads)
    _Float16* wpack = (_Float16*)d_ws;                         // PACK_TOTAL halves = 147456 B
    float* fbase = (float*)((char*)d_ws + PACK_TOTAL * sizeof(_Float16));
    float* h     = fbase;              // [N,64]
    float* AhB   = h   + NH;           // [N,64]
    float* BhB   = AhB + NH;           // [N,64]  (reused as tmp after edge pass)
    float* DhB   = BhB + NH;           // [N,64]
    float* EhB   = DhB + NH;           // [N,64]
    float* numB  = EhB + NH;           // [N,64]
    float* denB  = numB + NH;          // [N,64]
    float* stats = denB + NH;          // [128]
    float* tmp   = BhB;                // alias: Bh dead after edge_scatter

    const int row_tiles   = NNODES / 16;                 // 3125 (exact)
    const int gemm_blocks = (row_tiles + 7) / 8;         // 8 waves/block
    const int edge_blocks = (NEDGES + 7) / 8;            // 8 edges/block
    const int elem_blocks = (NH + 255) / 256;
    const int zero_elems  = 2 * NH + 128;                // num, den, stats contiguous
    const int zero_blocks = (zero_elems + 255) / 256;
    const int pack_blocks = (PACK_TOTAL + 255) / 256;

    pack_weights<<<pack_blocks, 256, 0, stream>>>(emb_w, Aw, Bw, Dw, Ew, wpack);

    embed_gemm<<<gemm_blocks, 256, 0, stream>>>(feature, wpack, emb_b, h);

    for (int l = 0; l < NLAYER; ++l) {
        zero_f32<<<zero_blocks, 256, 0, stream>>>(numB, zero_elems);

        layer_gemms<<<gemm_blocks, 256, 0, stream>>>(
            h,
            wpack + EMB_PACK + (size_t)l * 4 * MAT_PACK,
            Ab + (size_t)l * HID, Bb + (size_t)l * HID,
            Db + (size_t)l * HID, Eb + (size_t)l * HID,
            AhB, BhB, DhB, EhB);

        edge_scatter<<<edge_blocks, 256, 0, stream>>>(ei, BhB, DhB, EhB, numB, denB);

        bn_stats<<<1024, 256, 0, stream>>>(AhB, numB, denB, tmp, stats);

        float* dst = (l == NLAYER - 1) ? (float*)d_out : h;
        bn_apply<<<elem_blocks, 256, 0, stream>>>(h, tmp, stats,
                                                  gamma + (size_t)l * HID,
                                                  beta  + (size_t)l * HID, dst);
    }
}
